// TTT_DownProj_Wrapper_90434831384753
// MI455X (gfx1250) — compile-verified, matrix operand order checked by reading, not run on hardware
//
#include <hip/hip_runtime.h>
#include <hip/hip_bf16.h>

typedef __attribute__((ext_vector_type(16))) __bf16 v16bf;
typedef __attribute__((ext_vector_type(8)))  __bf16 v8bf;
typedef __attribute__((ext_vector_type(8)))  float  v8f;
typedef __attribute__((ext_vector_type(2)))  float  v2f;

#define K_DIM   1024
#define H_DIM   1024
#define SEQ     8192
#define BATCH   2
#define ROWS    (BATCH*SEQ)      /* 16384 */
#define CHUNK   256
#define NC      (SEQ/CHUNK)      /* 32 */
#define NBC     (BATCH*NC)       /* 64 */
#define RANK    16
#define SCALING_F 2.0f
#define LR_F      0.02f          /* TTT_LR * SCALING */
#define LN_EPS_F  1e-5f

// ---------------------------------------------------------------- helpers
__device__ __forceinline__ v8f wmma_bf16(v16bf a, v16bf b, v8f c) {
  // D = A(16x32) * B(32x16) + C, f32 accumulate
  return __builtin_amdgcn_wmma_f32_16x16x32_bf16(false, a, false, b, (short)0, c,
                                                 false, false);
}

// K index of element e of a 16-bit A/B fragment for this half-wave
__device__ __forceinline__ int kidx(int e, int hf) {
  return (e & 7) + hf * 8 + (e & 8) * 2;
}

// Load a 16x32 bf16 fragment row slice: p points at row start + k0 + hf*8
// Works for both global and LDS pointers (two 16B loads at +0 and +32B).
__device__ __forceinline__ v16bf load_frag(const __bf16* p) {
  const v8bf lo = *reinterpret_cast<const v8bf*>(p);
  const v8bf hi = *reinterpret_cast<const v8bf*>(p + 16);
  v16bf r;
#pragma unroll
  for (int e = 0; e < 8; ++e) { r[e] = lo[e]; r[e + 8] = hi[e]; }
  return r;
}

// ------------------------------------------- 1. LayerNorm + shift + bf16 x
__global__ void __launch_bounds__(256) prep_rows(
    const float* __restrict__ x, const float* __restrict__ gamma,
    const float* __restrict__ beta, __bf16* __restrict__ x_bf,
    __bf16* __restrict__ sh_bf) {
  __shared__ float red[256];
  const int s = blockIdx.x;
  const int t = threadIdx.x;
  const float* row = x + (size_t)s * K_DIM;
  float v[4];
  float sum = 0.f;
#pragma unroll
  for (int i = 0; i < 4; ++i) { v[i] = row[t + i * 256]; sum += v[i]; }
  red[t] = sum; __syncthreads();
  for (int off = 128; off; off >>= 1) {
    if (t < off) red[t] += red[t + off];
    __syncthreads();
  }
  const float mean = red[0] * (1.0f / K_DIM);
  __syncthreads();
  float ss = 0.f;
#pragma unroll
  for (int i = 0; i < 4; ++i) { float d = v[i] - mean; ss += d * d; }
  red[t] = ss; __syncthreads();
  for (int off = 128; off; off >>= 1) {
    if (t < off) red[t] += red[t + off];
    __syncthreads();
  }
  const float rstd = rsqrtf(red[0] * (1.0f / K_DIM) + LN_EPS_F);
  const int sb = s & (SEQ - 1);
#pragma unroll
  for (int i = 0; i < 4; ++i) {
    const int k = t + i * 256;
    x_bf[(size_t)s * K_DIM + k] = (__bf16)v[i];
    const float xn = (v[i] - mean) * rstd * gamma[k] + beta[k];
    if (sb > 0)        sh_bf[(size_t)(s - 1) * K_DIM + k] = (__bf16)xn;
    if (sb == SEQ - 1) sh_bf[(size_t)s * K_DIM + k] = (__bf16)0.0f;
  }
}

// ------------------------------------------- 2. weight conversions
__global__ void conv_weights(const float* __restrict__ Wb,
                             const float* __restrict__ Wp,
                             const float* __restrict__ iB,
                             const float* __restrict__ iA,
                             __bf16* __restrict__ Wb_bf,
                             __bf16* __restrict__ Wp_bf,
                             __bf16* __restrict__ iB_bf,
                             __bf16* __restrict__ iAT_bf) {
  const int NW = K_DIM * H_DIM;
  int t = blockIdx.x * blockDim.x + threadIdx.x;
  if (t < NW) {
    Wb_bf[t] = (__bf16)Wb[t];
  } else if (t < 2 * NW) {
    Wp_bf[t - NW] = (__bf16)Wp[t - NW];
  } else if (t < 2 * NW + RANK * K_DIM) {
    const int e = t - 2 * NW;
    iB_bf[e] = (__bf16)iB[e];
  } else if (t < 2 * NW + 2 * RANK * K_DIM) {
    const int e = t - 2 * NW - RANK * K_DIM;  // over init_A [H, r]
    const int h = e >> 4, r = e & 15;
    iAT_bf[r * K_DIM + h] = (__bf16)iA[e];    // transposed [r, H]
  }
}

// ------------------------------------------- 3. big GEMM  C = A * Bw^T
// A [ROWS,K] row-major bf16, Bw [H,K] row-major bf16.
// Block tile 128x128, 8 waves of 32x64. Double-buffered LDS staging:
// each 16KB K-stage (A 128x32 + B 128x32) is fetched from L2 exactly once
// per block, then fragments are served out of LDS (ds_load_b128).
#define BM 128
#define BN 128
#define BK 32
#define LDSS 40  /* padded row stride in bf16 elems (80B) to spread banks */

template <bool WF, bool WB>
__global__ void __launch_bounds__(256) gemm_bf16_nt(
    const __bf16* __restrict__ A, const __bf16* __restrict__ Bw,
    float* __restrict__ Cf, __bf16* __restrict__ Cbf) {
  constexpr int K = K_DIM;
  constexpr int N = H_DIM;
  __shared__ alignas(16) __bf16 sA[2][BM * LDSS];
  __shared__ alignas(16) __bf16 sB[2][BN * LDSS];

  const int t = threadIdx.x;
  const int lane = t & 31;
  const int wave = t >> 5;
  const int hf = lane >> 4;
  const int l16 = lane & 15;
  const int mblk = blockIdx.x * BM;
  const int nblk = blockIdx.y * BN;
  const int m0w = (wave & 3) * 32;   // wave tile row within block
  const int n0w = (wave >> 2) * 64;  // wave tile col within block

  // global staging: each thread owns rows r0 and r0+64, one 16B segment
  const int r0 = t >> 2;
  const int s0 = (t & 3) * 8;  // bf16 elems
  const __bf16* gA0 = A + (size_t)(mblk + r0) * K + s0;
  const __bf16* gA1 = A + (size_t)(mblk + r0 + 64) * K + s0;
  const __bf16* gB0 = Bw + (size_t)(nblk + r0) * K + s0;
  const __bf16* gB1 = Bw + (size_t)(nblk + r0 + 64) * K + s0;
  const int lo0 = r0 * LDSS + s0;
  const int lo1 = (r0 + 64) * LDSS + s0;

  v8f acc[2][4];
#pragma unroll
  for (int i = 0; i < 2; ++i)
#pragma unroll
    for (int j = 0; j < 4; ++j) acc[i][j] = {};

  // preload stage 0
  v8bf ra0 = *(const v8bf*)(gA0);
  v8bf ra1 = *(const v8bf*)(gA1);
  v8bf rb0 = *(const v8bf*)(gB0);
  v8bf rb1 = *(const v8bf*)(gB1);
  *(v8bf*)&sA[0][lo0] = ra0;
  *(v8bf*)&sA[0][lo1] = ra1;
  *(v8bf*)&sB[0][lo0] = rb0;
  *(v8bf*)&sB[0][lo1] = rb1;
  __syncthreads();

  constexpr int NS = K / BK;  // 32 stages
#pragma unroll 2
  for (int ks = 0; ks < NS; ++ks) {
    const int buf = ks & 1;
    if (ks + 1 < NS) {
      const int kn = (ks + 1) * BK;
      ra0 = *(const v8bf*)(gA0 + kn);
      ra1 = *(const v8bf*)(gA1 + kn);
      rb0 = *(const v8bf*)(gB0 + kn);
      rb1 = *(const v8bf*)(gB1 + kn);
    }
    v16bf a[2], b[4];
#pragma unroll
    for (int i = 0; i < 2; ++i)
      a[i] = load_frag(&sA[buf][(m0w + i * 16 + l16) * LDSS + hf * 8]);
#pragma unroll
    for (int j = 0; j < 4; ++j)
      b[j] = load_frag(&sB[buf][(n0w + j * 16 + l16) * LDSS + hf * 8]);
#pragma unroll
    for (int i = 0; i < 2; ++i)
#pragma unroll
      for (int j = 0; j < 4; ++j) acc[i][j] = wmma_bf16(a[i], b[j], acc[i][j]);
    if (ks + 1 < NS) {
      *(v8bf*)&sA[buf ^ 1][lo0] = ra0;
      *(v8bf*)&sA[buf ^ 1][lo1] = ra1;
      *(v8bf*)&sB[buf ^ 1][lo0] = rb0;
      *(v8bf*)&sB[buf ^ 1][lo1] = rb1;
    }
    __syncthreads();
  }

  // epilogue: one base pointer per 16x16 tile, stores use immediate offsets
#pragma unroll
  for (int i = 0; i < 2; ++i)
#pragma unroll
    for (int j = 0; j < 4; ++j) {
      const int row0 = mblk + m0w + i * 16 + hf * 8;
      const int col = nblk + n0w + j * 16 + l16;
      if (WF) {
        float* p = Cf + (size_t)row0 * N + col;
#pragma unroll
        for (int v = 0; v < 8; ++v) p[v * N] = acc[i][j][v];
      }
      if (WB) {
        __bf16* p = Cbf + (size_t)row0 * N + col;
#pragma unroll
        for (int v = 0; v < 8; ++v) p[v * N] = (__bf16)acc[i][j][v];
      }
    }
}

// ------------------------------------------- 4a. proj_in / proj_err
// wave0: proj_in = Z @ init_B^T ; wave1: proj_err = V @ init_A
__global__ void __launch_bounds__(64) proj_kernel(
    const __bf16* __restrict__ x_bf, const __bf16* __restrict__ V_bf,
    const __bf16* __restrict__ iB_bf, const __bf16* __restrict__ iAT_bf,
    float* __restrict__ proj_in, float* __restrict__ proj_err) {
  const int bc = blockIdx.x, mt = blockIdx.y;
  const int wave = threadIdx.x >> 5;
  const int lane = threadIdx.x & 31;
  const int hf = lane >> 4, l16 = lane & 15;
  const __bf16* Amat = wave ? V_bf : x_bf;
  const __bf16* Bw = wave ? iAT_bf : iB_bf;
  float* outp = wave ? proj_err : proj_in;
  const int row0 = bc * CHUNK + mt * 16;
  const __bf16* pa = Amat + (size_t)(row0 + l16) * K_DIM + hf * 8;
  const __bf16* pb = Bw + (size_t)l16 * K_DIM + hf * 8;
  v8f acc = {};
  for (int k0 = 0; k0 < K_DIM; k0 += 32) {
    acc = wmma_bf16(load_frag(pa + k0), load_frag(pb + k0), acc);
  }
  float* p = outp + (size_t)bc * CHUNK * RANK + (mt * 16 + hf * 8) * RANK + l16;
#pragma unroll
  for (int v = 0; v < 8; ++v) p[v * RANK] = acc[v];
}

// ------------------------------------------- 4b. dA = Vc^T @ proj_in
__global__ void __launch_bounds__(32) dA_kernel(
    const __bf16* __restrict__ V_bf, const float* __restrict__ proj_in,
    float* __restrict__ dA) {
  const int bc = blockIdx.x, ht = blockIdx.y;
  const int lane = threadIdx.x & 31;
  const int hf = lane >> 4, l16 = lane & 15;
  const int h0 = ht * 16;
  const float* pi = proj_in + (size_t)bc * CHUNK * RANK;
  v8f acc = {};
  for (int c0 = 0; c0 < CHUNK; c0 += 32) {
    v16bf a, b;
#pragma unroll
    for (int e = 0; e < 16; ++e) {
      const int kk = c0 + kidx(e, hf);
      a[e] = V_bf[(size_t)(bc * CHUNK + kk) * K_DIM + h0 + l16];
      b[e] = (__bf16)pi[kk * RANK + l16];
    }
    acc = wmma_bf16(a, b, acc);
  }
  float* p = dA + (size_t)bc * H_DIM * RANK + (h0 + hf * 8) * RANK + l16;
#pragma unroll
  for (int v = 0; v < 8; ++v) p[v * RANK] = acc[v];
}

// ------------------------------------------- 4c. dB = proj_err^T @ Z
__global__ void __launch_bounds__(32) dB_kernel(
    const float* __restrict__ proj_err, const __bf16* __restrict__ x_bf,
    float* __restrict__ dB) {
  const int bc = blockIdx.x, nt = blockIdx.y;
  const int lane = threadIdx.x & 31;
  const int hf = lane >> 4, l16 = lane & 15;
  const int n0 = nt * 16;
  const float* pe = proj_err + (size_t)bc * CHUNK * RANK;
  v8f acc = {};
  for (int c0 = 0; c0 < CHUNK; c0 += 32) {
    v16bf a, b;
#pragma unroll
    for (int e = 0; e < 16; ++e) {
      const int kk = c0 + kidx(e, hf);
      a[e] = (__bf16)pe[kk * RANK + l16];
      b[e] = x_bf[(size_t)(bc * CHUNK + kk) * K_DIM + n0 + l16];
    }
    acc = wmma_bf16(a, b, acc);
  }
  float* p = dB + (size_t)bc * RANK * K_DIM + (hf * 8) * K_DIM + n0 + l16;
#pragma unroll
  for (int v = 0; v < 8; ++v) p[v * K_DIM] = acc[v];
}

// ------------------------------------------- 5a. exclusive cumsum * lr (in place)
__global__ void cumsum_kernel(float* __restrict__ dA, float* __restrict__ dB) {
  const int PER = H_DIM * RANK;  // 16384
  int t = blockIdx.x * blockDim.x + threadIdx.x;  // [0, 4*PER)
  float* buf = (t < 2 * PER) ? dA : dB;
  const int tt = (t < 2 * PER) ? t : t - 2 * PER;
  const int b = tt / PER;
  const int e = tt % PER;
  float acc = 0.f;
  for (int c = 0; c < NC; ++c) {
    const size_t idx = ((size_t)(b * NC + c)) * PER + e;
    const float old = buf[idx];
    buf[idx] = LR_F * acc;
    acc += old;
  }
}

// ------------------------------------------- 5b. joint norm clip coefficient
__global__ void __launch_bounds__(256) norm_kernel(
    const float* __restrict__ rawA, const float* __restrict__ rawB,
    float* __restrict__ coef) {
  __shared__ float red[256];
  const int bc = blockIdx.x, t = threadIdx.x;
  const int PER = H_DIM * RANK;
  float s = 0.f;
  for (int i = t; i < PER; i += 256) { float v = rawA[(size_t)bc * PER + i]; s += v * v; }
  for (int i = t; i < PER; i += 256) { float v = rawB[(size_t)bc * PER + i]; s += v * v; }
  red[t] = s; __syncthreads();
  for (int off = 128; off; off >>= 1) {
    if (t < off) red[t] += red[t + off];
    __syncthreads();
  }
  if (t == 0) {
    const float n = sqrtf(red[0]);
    coef[bc] = fminf(1.0f / (n + 1e-6f), 1.0f);
  }
}

// ------------------------------------------- 5c. effective weights
__global__ void eff_kernel(const float* __restrict__ iA,
                           const float* __restrict__ iB,
                           const float* __restrict__ rawA,
                           const float* __restrict__ rawB,
                           const float* __restrict__ coef,
                           float* __restrict__ A_eff,
                           __bf16* __restrict__ B_eff_bf) {
  const int PER = H_DIM * RANK;
  const int TOT = NBC * PER;
  int t = blockIdx.x * blockDim.x + threadIdx.x;
  if (t < TOT) {
    const int bc = t / PER, e = t % PER;
    A_eff[t] = iA[e] - rawA[t] * coef[bc];
  } else if (t < 2 * TOT) {
    const int tt = t - TOT;
    const int bc = tt / PER, e = tt % PER;
    B_eff_bf[tt] = (__bf16)(iB[e] - rawB[tt] * coef[bc]);
  }
}

// ------------------------------------------- 6a. mid = Z @ B_eff^T
__global__ void __launch_bounds__(32) mid_kernel(
    const __bf16* __restrict__ x_bf, const __bf16* __restrict__ B_eff_bf,
    float* __restrict__ mid) {
  const int bc = blockIdx.x, mt = blockIdx.y;
  const int lane = threadIdx.x & 31;
  const int hf = lane >> 4, l16 = lane & 15;
  const int row0 = bc * CHUNK + mt * 16;
  const __bf16* pa = x_bf + (size_t)(row0 + l16) * K_DIM + hf * 8;
  const __bf16* pb = B_eff_bf + (size_t)bc * RANK * K_DIM + (size_t)l16 * K_DIM + hf * 8;
  v8f acc = {};
  for (int k0 = 0; k0 < K_DIM; k0 += 32) {
    acc = wmma_bf16(load_frag(pa + k0), load_frag(pb + k0), acc);
  }
  float* p = mid + (size_t)bc * CHUNK * RANK + (mt * 16 + hf * 8) * RANK + l16;
#pragma unroll
  for (int v = 0; v < 8; ++v) p[v * RANK] = acc[v];
}

// ------------------------------------------- 6b. out += SCALING * mid @ A_eff^T
__global__ void __launch_bounds__(256) final_kernel(
    const float* __restrict__ mid, const float* __restrict__ A_eff,
    float* __restrict__ out) {
  const int bc = blockIdx.x, mt = blockIdx.y;
  const int wave = threadIdx.x >> 5;
  const int nt = blockIdx.z * 8 + wave;
  const int lane = threadIdx.x & 31;
  const int hf = lane >> 4, l16 = lane & 15;
  const float* Mp = mid + (size_t)bc * CHUNK * RANK + (size_t)(mt * 16) * RANK;
  const float* Ae = A_eff + (size_t)bc * H_DIM * RANK + (size_t)(nt * 16) * RANK;
  v8f acc = {};
#if __has_builtin(__builtin_amdgcn_wmma_f32_16x16x4_f32)
#pragma unroll
  for (int k0 = 0; k0 < RANK; k0 += 4) {
    v2f a, b;
    a[0] = Mp[l16 * RANK + k0 + hf * 2 + 0];
    a[1] = Mp[l16 * RANK + k0 + hf * 2 + 1];
    b[0] = Ae[l16 * RANK + k0 + hf * 2 + 0];
    b[1] = Ae[l16 * RANK + k0 + hf * 2 + 1];
    acc = __builtin_amdgcn_wmma_f32_16x16x4_f32(false, a, false, b, (short)0,
                                                acc, false, false);
  }
#else
#pragma unroll
  for (int v = 0; v < 8; ++v) {
    const int rloc = v + hf * 8;
    float s = 0.f;
    for (int k = 0; k < RANK; ++k) s += Mp[rloc * RANK + k] * Ae[l16 * RANK + k];
    acc[v] = s;
  }
#endif
  float* p = out + (size_t)(bc * CHUNK + mt * 16 + hf * 8) * H_DIM + nt * 16 + l16;
#pragma unroll
  for (int v = 0; v < 8; ++v) p[v * H_DIM] += SCALING_F * acc[v];
}

// ---------------------------------------------------------------- launch
extern "C" void kernel_launch(void* const* d_in, const int* in_sizes, int n_in,
                              void* d_out, int out_size, void* d_ws,
                              size_t ws_size, hipStream_t stream) {
  (void)in_sizes; (void)n_in; (void)out_size; (void)ws_size;
  const float* x  = (const float*)d_in[0];
  const float* Wb = (const float*)d_in[1];
  const float* iA = (const float*)d_in[2];
  const float* iB = (const float*)d_in[3];
  const float* g  = (const float*)d_in[4];
  const float* be = (const float*)d_in[5];
  const float* Wp = (const float*)d_in[6];
  float* out = (float*)d_out;

  char* ws = (char*)d_ws;
  size_t off = 0;
  auto alloc = [&](size_t bytes) {
    void* p = ws + off;
    off += (bytes + 255) & ~(size_t)255;
    return p;
  };
  __bf16* x_bf     = (__bf16*)alloc((size_t)ROWS * K_DIM * 2);
  __bf16* sh_bf    = (__bf16*)alloc((size_t)ROWS * K_DIM * 2);
  __bf16* V_bf     = (__bf16*)alloc((size_t)ROWS * K_DIM * 2);
  __bf16* Wb_bf    = (__bf16*)alloc((size_t)H_DIM * K_DIM * 2);
  __bf16* Wp_bf    = (__bf16*)alloc((size_t)H_DIM * K_DIM * 2);
  __bf16* iB_bf    = (__bf16*)alloc((size_t)RANK * K_DIM * 2);
  __bf16* iAT_bf   = (__bf16*)alloc((size_t)RANK * K_DIM * 2);
  float*  proj_in  = (float*)alloc((size_t)NBC * CHUNK * RANK * 4);
  float*  proj_err = (float*)alloc((size_t)NBC * CHUNK * RANK * 4);
  float*  dAbuf    = (float*)alloc((size_t)NBC * H_DIM * RANK * 4);
  float*  dBbuf    = (float*)alloc((size_t)NBC * RANK * K_DIM * 4);
  float*  coef     = (float*)alloc((size_t)NBC * 4);
  float*  A_eff    = (float*)alloc((size_t)NBC * H_DIM * RANK * 4);
  __bf16* B_eff_bf = (__bf16*)alloc((size_t)NBC * RANK * K_DIM * 2);
  float*  midbuf   = (float*)alloc((size_t)NBC * CHUNK * RANK * 4);

  // 1. LayerNorm + shift targets + bf16 conversion of x
  prep_rows<<<ROWS, 256, 0, stream>>>(x, g, be, x_bf, sh_bf);

  // 2. weight conversions
  const int convN = 2 * K_DIM * H_DIM + 2 * RANK * K_DIM;
  conv_weights<<<(convN + 255) / 256, 256, 0, stream>>>(
      Wb, Wp, iB, iA, Wb_bf, Wp_bf, iB_bf, iAT_bf);

  // 3. base_out = x @ Wb^T -> d_out ; V = shifted @ Wp^T -> V_bf
  dim3 gg(ROWS / BM, H_DIM / BN);
  gemm_bf16_nt<true, false><<<gg, 256, 0, stream>>>(x_bf, Wb_bf, out, nullptr);
  gemm_bf16_nt<false, true><<<gg, 256, 0, stream>>>(sh_bf, Wp_bf, nullptr, V_bf);

  // 4. per-chunk fast-weight deltas
  proj_kernel<<<dim3(NBC, CHUNK / 16), 64, 0, stream>>>(
      x_bf, V_bf, iB_bf, iAT_bf, proj_in, proj_err);
  dA_kernel<<<dim3(NBC, H_DIM / 16), 32, 0, stream>>>(V_bf, proj_in, dAbuf);
  dB_kernel<<<dim3(NBC, K_DIM / 16), 32, 0, stream>>>(proj_err, x_bf, dBbuf);

  // 5. exclusive cumsum * lr, joint norm clip, effective weights
  cumsum_kernel<<<(4 * H_DIM * RANK) / 256, 256, 0, stream>>>(dAbuf, dBbuf);
  norm_kernel<<<NBC, 256, 0, stream>>>(dAbuf, dBbuf, coef);
  eff_kernel<<<(2 * NBC * H_DIM * RANK) / 256, 256, 0, stream>>>(
      iA, iB, dAbuf, dBbuf, coef, A_eff, B_eff_bf);

  // 6. LoRA apply
  mid_kernel<<<dim3(NBC, CHUNK / 16), 32, 0, stream>>>(x_bf, B_eff_bf, midbuf);
  final_kernel<<<dim3(NBC, CHUNK / 16, H_DIM / 128), 256, 0, stream>>>(
      midbuf, A_eff, out);
}
